// SelfAttention_75763223101718
// MI455X (gfx1250) — matrix operand order repaired and checked by reading, not compiled, not run on hardware
//
#include <hip/hip_runtime.h>

// ---------------- problem constants ----------------
#define H_  1024
#define NH_ 16
#define HD_ 64
#define B_  4
#define S_  2048
#define M_  (B_ * S_)   // 8192 rows of X
#define NBLK_ (S_ / 32) // 64 key blocks

// ---------------- types ----------------
typedef __attribute__((ext_vector_type(16))) __bf16 v16bf;
typedef __attribute__((ext_vector_type(8)))  float  v8f;
typedef __attribute__((ext_vector_type(4)))  unsigned int u32x4;
typedef __attribute__((ext_vector_type(8)))  int i32x8;
typedef __attribute__((ext_vector_type(4)))  int i32x4;

union Frag { v16bf v; uint4 u[2]; };

static __device__ __forceinline__ unsigned short f32_to_bf16(float f) {
  unsigned int u = __float_as_uint(f);
  u += 0x7FFFu + ((u >> 16) & 1u);            // round-to-nearest-even
  return (unsigned short)(u >> 16);
}

static __device__ __forceinline__ v8f wmma_bf16(v16bf a, v16bf b, v8f c) {
  // (neg_a, A, neg_b, B, c_mod, C, reuse_a, reuse_b)
  return __builtin_amdgcn_wmma_f32_16x16x32_bf16(false, a, false, b, (short)0, c,
                                                 false, false);
}

// ---------------- TDM (Tensor Data Mover) staging ----------------
#ifndef ATHENA_USE_TDM
#  if defined(__has_builtin)
#    if __has_builtin(__builtin_amdgcn_tensor_load_to_lds)
#      define ATHENA_USE_TDM 1
#    else
#      define ATHENA_USE_TDM 0
#    endif
#  else
#    define ATHENA_USE_TDM 0
#  endif
#endif

static __device__ __forceinline__ void wait_tensorcnt0() {
#if defined(__has_builtin) && __has_builtin(__builtin_amdgcn_s_wait_tensorcnt)
  __builtin_amdgcn_s_wait_tensorcnt(0);
#else
  asm volatile("s_wait_tensorcnt 0x0" ::: "memory");
#endif
}

static __device__ __forceinline__ void wait_tensorcnt2() {
#if defined(__has_builtin) && __has_builtin(__builtin_amdgcn_s_wait_tensorcnt)
  __builtin_amdgcn_s_wait_tensorcnt(2);
#else
  asm volatile("s_wait_tensorcnt 0x2" ::: "memory");
#endif
}

#if ATHENA_USE_TDM
// 2D tile load, 2-byte elements. Dimensions in elements.
static __device__ __forceinline__ void tdm_load_2d(
    unsigned lds_off, const unsigned short* gptr,
    unsigned tensor_d0, unsigned tensor_d1,
    unsigned tile_d0, unsigned tile_d1, unsigned stride0)
{
  unsigned long long ga = (unsigned long long)(size_t)gptr;
  u32x4 g0 = { 1u,                                    // count=1, user mode
               lds_off,                               // lds_addr (bytes)
               (unsigned)(ga & 0xFFFFFFFFull),        // global_addr[31:0]
               (unsigned)((ga >> 32) & 0x1FFFFFFull) | 0x80000000u }; // addr[56:32] | type=2
  i32x8 g1 = { (int)0x00010000u,                                   // data_size=1 (2B)
               (int)((tensor_d0 & 0xFFFFu) << 16),                 // dim0[15:0]
               (int)(((tensor_d0 >> 16) & 0xFFFFu) |
                     ((tensor_d1 & 0xFFFFu) << 16)),               // dim0 hi | dim1 lo
               (int)(((tensor_d1 >> 16) & 0xFFFFu) |
                     ((tile_d0 & 0xFFFFu) << 16)),                 // dim1 hi | tile0
               (int)(tile_d1 & 0xFFFFu),                           // tile1 (tile2=0)
               (int)stride0,                                       // dim0_stride lo
               0, 0 };
  i32x4 gz = { 0, 0, 0, 0 };
#if defined(__clang_major__) && (__clang_major__ >= 23)
  i32x8 gp = { 0, 0, 0, 0, 0, 0, 0, 0 };
  __builtin_amdgcn_tensor_load_to_lds(g0, g1, gz, gz, gp, 0);
#else
  __builtin_amdgcn_tensor_load_to_lds(g0, g1, gz, gz, 0);
#endif
}
#endif

// ---------------- kernel 1: f32 -> bf16 convert ----------------
__global__ void __launch_bounds__(256)
cvt_f32_bf16_kernel(const float* __restrict__ src, unsigned short* __restrict__ dst, int n)
{
  int i = blockIdx.x * 256 + threadIdx.x;
  int stride = gridDim.x * 256;
  for (; i < n; i += stride) dst[i] = f32_to_bf16(src[i]);
}

// ---------------- kernel 2: fused QKV projection ----------------
// out = X @ W^T + b ; grid (H/64, M/128, 3), block 256 (8 waves, 16 rows each)
__global__ void __launch_bounds__(256)
gemm_qkv_kernel(const unsigned short* __restrict__ Xb,
                const unsigned short* __restrict__ Wqb,
                const unsigned short* __restrict__ Wkb,
                const unsigned short* __restrict__ Wvb,
                const float* __restrict__ bq, const float* __restrict__ bk,
                const float* __restrict__ bv,
                unsigned short* __restrict__ Qb, unsigned short* __restrict__ Kb,
                unsigned short* __restrict__ Vtb)
{
  const int lane = threadIdx.x & 31;
  const int wave = threadIdx.x >> 5;
  const int c    = lane & 15;
  const int kh8  = (lane >> 4) << 3;   // unused; A and B loads use kh16
  const int kh16 = (lane >> 4) << 4;   // contiguous K split (k = 16h + s) for A and B
  const int n0 = blockIdx.x * 64;
  const int m0 = blockIdx.y * 128 + wave * 16;
  const int z  = blockIdx.z;
  const unsigned short* Wb = (z == 0) ? Wqb : (z == 1) ? Wkb : Wvb;
  const float* bias = (z == 0) ? bq : (z == 1) ? bk : bv;

  const unsigned short* arow = Xb + (size_t)(m0 + c) * H_;
  v8f acc[4] = {};

  for (int k0 = 0; k0 < H_; k0 += 32) {
    Frag a, bf[4];
    a.u[0] = *(const uint4*)(arow + k0 + kh16);
    a.u[1] = *(const uint4*)(arow + k0 + kh16 + 8);
    __builtin_prefetch(arow + k0 + 64, 0, 1);  // global_prefetch_b8
#pragma unroll
    for (int nt = 0; nt < 4; ++nt) {
      const unsigned short* bp = Wb + (size_t)(n0 + nt * 16 + c) * H_ + k0 + kh16;
      bf[nt].u[0] = *(const uint4*)(bp);
      bf[nt].u[1] = *(const uint4*)(bp + 8);
    }
#pragma unroll
    for (int nt = 0; nt < 4; ++nt)
      acc[nt] = wmma_bf16(a.v, bf[nt].v, acc[nt]);
  }

  const int rbase = m0 + ((lane >> 4) << 3);
#pragma unroll
  for (int nt = 0; nt < 4; ++nt) {
    const int ncol = n0 + nt * 16 + c;
    const float bb  = bias[ncol];
    const int h = ncol >> 6, d = ncol & 63;
#pragma unroll
    for (int i = 0; i < 8; ++i) {
      float v = acc[nt][i] + bb;
      int m = rbase + i;
      int bidx = m >> 11;         // / S_
      int s    = m & (S_ - 1);
      if (z == 0) {
        Qb[(((size_t)bidx * NH_ + h) * S_ + s) * HD_ + d] = f32_to_bf16(v * 0.125f);
      } else if (z == 1) {
        Kb[(((size_t)bidx * NH_ + h) * S_ + s) * HD_ + d] = f32_to_bf16(v);
      } else { // V stored transposed: [b][h][d][s]
        Vtb[(((size_t)bidx * NH_ + h) * HD_ + d) * S_ + s] = f32_to_bf16(v);
      }
    }
  }
}

// ---------------- kernel 3: flash attention ----------------
// grid (S/128, B*NH), block 256 (8 waves x 16 q-rows)
// K/V tiles double-buffered in LDS; TDM for tile i+1 issued before computing
// tile i so the DMA is hidden behind 8 WMMAs + softmax VALU work per block.
__global__ void __launch_bounds__(256)
attn_kernel(const unsigned short* __restrict__ Qb,
            const unsigned short* __restrict__ Kb,
            const unsigned short* __restrict__ Vtb,
            const float* __restrict__ maskp,
            unsigned short* __restrict__ Ctxb)
{
  __shared__ __align__(16) unsigned short Ktile[2][32 * HD_];   // [buf][key][d]
  __shared__ __align__(16) unsigned short Vtile[2][HD_ * 32];   // [buf][d][key]
  __shared__ __align__(16) unsigned short Ptile[8][16 * 32];    // per-wave P

  const int tid  = threadIdx.x;
  const int lane = tid & 31;
  const int wave = tid >> 5;
  const int c    = lane & 15;
  const int kh8  = (lane >> 4) << 3;
  const int kh16 = (lane >> 4) << 4;
  const int bh = blockIdx.y;
  const int b  = bh >> 4;          // NH_ == 16
  const int hh = bh & 15;
  const int m0 = blockIdx.x * 128 + wave * 16;

  const unsigned short* kbase = Kb  + (size_t)bh * S_ * HD_;
  const unsigned short* vbase = Vtb + (size_t)bh * HD_ * S_;
  const float* mrow = maskp + (size_t)b * S_;

  // Q fragments (A layout), rows m0..m0+15, d split in two K=32 chunks
  const unsigned short* qrow = Qb + ((size_t)bh * S_ + (m0 + c)) * HD_;
  Frag aq0, aq1;
  aq0.u[0] = *(const uint4*)(qrow + kh16);
  aq0.u[1] = *(const uint4*)(qrow + kh16 + 8);
  aq1.u[0] = *(const uint4*)(qrow + 32 + kh16);
  aq1.u[1] = *(const uint4*)(qrow + 32 + kh16 + 8);

  float Mrow[8], Lrow[8];
  v8f cacc[4] = {};
#pragma unroll
  for (int i = 0; i < 8; ++i) { Mrow[i] = -3.0e38f; Lrow[i] = 0.0f; }

#if ATHENA_USE_TDM
  // prologue: kick off tile 0 DMA
  if (wave == 0) {
    tdm_load_2d((unsigned)(size_t)&Ktile[0][0], kbase, HD_, S_, HD_, 32, HD_);
    tdm_load_2d((unsigned)(size_t)&Vtile[0][0], vbase, S_, HD_, 32, HD_, S_);
  }
#endif

  for (int it = 0; it < NBLK_; ++it) {
    const int s0  = it * 32;
    const int buf = it & 1;
    __syncthreads();  // all waves done reading buf^1 (iteration it-1)
#if ATHENA_USE_TDM
    if (wave == 0) {
      if (it + 1 < NBLK_) {
        tdm_load_2d((unsigned)(size_t)&Ktile[buf ^ 1][0],
                    kbase + (size_t)(s0 + 32) * HD_, HD_, S_, HD_, 32, HD_);
        tdm_load_2d((unsigned)(size_t)&Vtile[buf ^ 1][0],
                    vbase + (s0 + 32), S_, HD_, 32, HD_, S_);
        wait_tensorcnt2();   // tile `it` complete (TDM in-order per wave)
      } else {
        wait_tensorcnt0();
      }
    }
#else
    {
      ((uint4*)&Ktile[buf][0])[tid] =
          *(const uint4*)(kbase + (size_t)s0 * HD_ + tid * 8);
      int d = tid >> 2, cc = (tid & 3) * 8;
      *(uint4*)&Vtile[buf][d * 32 + cc] =
          *(const uint4*)(vbase + (size_t)d * S_ + s0 + cc);
    }
#endif
    __syncthreads();  // tile `it` visible to all waves

    // ---- scores: Q(16x64) x K^T(64x32) -> 2 n-tiles ----
    v8f sc[2] = {};
#pragma unroll
    for (int nt = 0; nt < 2; ++nt) {
      const int j = nt * 16 + c;            // key (B-matrix column) for this lane
      Frag bk0, bk1;
      bk0.u[0] = *(const uint4*)&Ktile[buf][j * HD_ + kh16];
      bk0.u[1] = *(const uint4*)&Ktile[buf][j * HD_ + kh16 + 8];
      bk1.u[0] = *(const uint4*)&Ktile[buf][j * HD_ + 32 + kh16];
      bk1.u[1] = *(const uint4*)&Ktile[buf][j * HD_ + 32 + kh16 + 8];
      sc[nt] = wmma_bf16(aq0.v, bk0.v, sc[nt]);
      sc[nt] = wmma_bf16(aq1.v, bk1.v, sc[nt]);
    }

    const float mk0 = mrow[s0 + c];
    const float mk1 = mrow[s0 + 16 + c];
    const int rl = (lane >> 4) << 3;

    // ---- online softmax (per row = per accumulator index i) ----
#pragma unroll
    for (int i = 0; i < 8; ++i) {
      float v0 = sc[0][i] + mk0;
      float v1 = sc[1][i] + mk1;
      float mx = fmaxf(v0, v1);
#pragma unroll
      for (int off = 8; off >= 1; off >>= 1)
        mx = fmaxf(mx, __shfl_xor(mx, off, 32));
      float mnew = fmaxf(Mrow[i], mx);
      float corr = __expf(Mrow[i] - mnew);
      float p0 = __expf(v0 - mnew);
      float p1 = __expf(v1 - mnew);
      float ps = p0 + p1;
#pragma unroll
      for (int off = 8; off >= 1; off >>= 1)
        ps += __shfl_xor(ps, off, 32);
      Lrow[i] = Lrow[i] * corr + ps;
      Mrow[i] = mnew;
      cacc[0][i] *= corr; cacc[1][i] *= corr;
      cacc[2][i] *= corr; cacc[3][i] *= corr;
      // store P (C layout -> LDS), re-read below in A layout (same-wave, in-order)
      Ptile[wave][(rl + i) * 32 + c]      = f32_to_bf16(p0);
      Ptile[wave][(rl + i) * 32 + 16 + c] = f32_to_bf16(p1);
    }

    // ---- ctx += P(16x32) x V(32x64) ----
    Frag pf;
    pf.u[0] = *(const uint4*)&Ptile[wave][c * 32 + kh16];
    pf.u[1] = *(const uint4*)&Ptile[wave][c * 32 + kh16 + 8];
#pragma unroll
    for (int nt = 0; nt < 4; ++nt) {
      const int d = nt * 16 + c;            // output column for this lane
      Frag bv;
      bv.u[0] = *(const uint4*)&Vtile[buf][d * 32 + kh16];
      bv.u[1] = *(const uint4*)&Vtile[buf][d * 32 + kh16 + 8];
      cacc[nt] = wmma_bf16(pf.v, bv.v, cacc[nt]);
    }
  }

  // ---- finalize: ctx / L, store bf16 in [b][s][h*HD+d] for the out-proj ----
  const int rbase = m0 + ((lane >> 4) << 3);
#pragma unroll
  for (int i = 0; i < 8; ++i) {
    float inv = 1.0f / Lrow[i];
    size_t obase = ((size_t)b * S_ + (rbase + i)) * H_ + hh * HD_;
#pragma unroll
    for (int nt = 0; nt < 4; ++nt)
      Ctxb[obase + nt * 16 + c] = f32_to_bf16(cacc[nt][i] * inv);
  }
}

// ---------------- kernel 4: output projection (f32 out) ----------------
__global__ void __launch_bounds__(256)
gemm_out_kernel(const unsigned short* __restrict__ Ab,
                const unsigned short* __restrict__ Wb,
                const float* __restrict__ bias, float* __restrict__ out)
{
  const int lane = threadIdx.x & 31;
  const int wave = threadIdx.x >> 5;
  const int c    = lane & 15;
  const int kh8  = (lane >> 4) << 3;
  const int kh16 = (lane >> 4) << 4;
  const int n0 = blockIdx.x * 64;
  const int m0 = blockIdx.y * 128 + wave * 16;

  const unsigned short* arow = Ab + (size_t)(m0 + c) * H_;
  v8f acc[4] = {};
  for (int k0 = 0; k0 < H_; k0 += 32) {
    Frag a, bf[4];
    a.u[0] = *(const uint4*)(arow + k0 + kh16);
    a.u[1] = *(const uint4*)(arow + k0 + kh16 + 8);
    __builtin_prefetch(arow + k0 + 64, 0, 1);
#pragma unroll
    for (int nt = 0; nt < 4; ++nt) {
      const unsigned short* bp = Wb + (size_t)(n0 + nt * 16 + c) * H_ + k0 + kh16;
      bf[nt].u[0] = *(const uint4*)(bp);
      bf[nt].u[1] = *(const uint4*)(bp + 8);
    }
#pragma unroll
    for (int nt = 0; nt < 4; ++nt)
      acc[nt] = wmma_bf16(a.v, bf[nt].v, acc[nt]);
  }
  const int rbase = m0 + ((lane >> 4) << 3);
#pragma unroll
  for (int nt = 0; nt < 4; ++nt) {
    const int ncol = n0 + nt * 16 + c;
    const float bb = bias[ncol];
#pragma unroll
    for (int i = 0; i < 8; ++i)
      out[(size_t)(rbase + i) * H_ + ncol] = acc[nt][i] + bb;
  }
}

// ---------------- launch ----------------
extern "C" void kernel_launch(void* const* d_in, const int* in_sizes, int n_in,
                              void* d_out, int out_size, void* d_ws, size_t ws_size,
                              hipStream_t stream) {
  const float* X    = (const float*)d_in[0];
  const float* mask = (const float*)d_in[1];
  const float* Wq   = (const float*)d_in[2];
  const float* bq   = (const float*)d_in[3];
  const float* Wk   = (const float*)d_in[4];
  const float* bk   = (const float*)d_in[5];
  const float* Wv   = (const float*)d_in[6];
  const float* bv   = (const float*)d_in[7];
  const float* Wo   = (const float*)d_in[8];
  const float* bo   = (const float*)d_in[9];
  float* out = (float*)d_out;

  // workspace layout (bytes); total = 88 MiB
  char* ws = (char*)d_ws;
  const size_t SZ_X = (size_t)M_ * H_ * 2;   // 16 MiB
  const size_t SZ_W = (size_t)H_ * H_ * 2;   // 2 MiB
  const size_t SZ_Q = (size_t)M_ * H_ * 2;   // 16 MiB
  unsigned short* Xb   = (unsigned short*)(ws);
  unsigned short* Wqb  = (unsigned short*)(ws + SZ_X);
  unsigned short* Wkb  = (unsigned short*)(ws + SZ_X + SZ_W);
  unsigned short* Wvb  = (unsigned short*)(ws + SZ_X + 2 * SZ_W);
  unsigned short* Wob  = (unsigned short*)(ws + SZ_X + 3 * SZ_W);
  unsigned short* Qb   = (unsigned short*)(ws + SZ_X + 4 * SZ_W);
  unsigned short* Kb   = (unsigned short*)(ws + SZ_X + 4 * SZ_W + SZ_Q);
  unsigned short* Vtb  = (unsigned short*)(ws + SZ_X + 4 * SZ_W + 2 * SZ_Q);
  unsigned short* Ctxb = (unsigned short*)(ws + SZ_X + 4 * SZ_W + 3 * SZ_Q);

  cvt_f32_bf16_kernel<<<2048, 256, 0, stream>>>(X,  Xb,  M_ * H_);
  cvt_f32_bf16_kernel<<<1024, 256, 0, stream>>>(Wq, Wqb, H_ * H_);
  cvt_f32_bf16_kernel<<<1024, 256, 0, stream>>>(Wk, Wkb, H_ * H_);
  cvt_f32_bf16_kernel<<<1024, 256, 0, stream>>>(Wv, Wvb, H_ * H_);
  cvt_f32_bf16_kernel<<<1024, 256, 0, stream>>>(Wo, Wob, H_ * H_);

  gemm_qkv_kernel<<<dim3(H_ / 64, M_ / 128, 3), 256, 0, stream>>>(
      Xb, Wqb, Wkb, Wvb, bq, bk, bv, Qb, Kb, Vtb);

  attn_kernel<<<dim3(S_ / 128, B_ * NH_), 256, 0, stream>>>(
      Qb, Kb, Vtb, mask, Ctxb);

  gemm_out_kernel<<<dim3(H_ / 64, M_ / 128), 256, 0, stream>>>(
      Ctxb, Wob, bo, out);
}
